// QuantumNeuralNetwork_35639638622877
// MI455X (gfx1250) — matrix operand order repaired and checked by reading, not compiled, not run on hardware
//
#include <hip/hip_runtime.h>

typedef __attribute__((ext_vector_type(16))) _Float16 v16h;
typedef __attribute__((ext_vector_type(8)))  _Float16 v8h;
typedef __attribute__((ext_vector_type(8)))  float    v8f;
typedef __attribute__((ext_vector_type(4)))  float    f32x4;

#define QNN_BLOCK 256   // 8 wave32s; each wave owns 32 samples (two 16-row WMMA M-tiles)

__device__ __forceinline__ v8f wmma_f16(v16h a, v16h b, v8f c) {
  // D = A(16x32 f16) * B(32x16 f16) + C(16x16 f32)
  return __builtin_amdgcn_wmma_f32_16x16x32_f16(false, a, false, b, (short)0, c, false, false);
}

__device__ __forceinline__ v16h cat16(v8h lo, v8h hi) {
  return __builtin_shufflevector(lo, hi, 0,1,2,3,4,5,6,7,8,9,10,11,12,13,14,15);
}

__device__ __forceinline__ v8h cvt8(f32x4 a, f32x4 b) {
  v8f f = __builtin_shufflevector(a, b, 0,1,2,3,4,5,6,7);
  return __builtin_convertvector(f, v8h);
}

// ---------------- A fragments (lane&15 = M row, lane>>4 = K-half kh) ----------------
// v16h element e: e<8 -> k = kh*8 + e ; e>=8 -> k = 16 + kh*8 + (e-8)   (+ 32*chunk)

// text: K=16 real, k==16 carries the bias-1 (lane-half 0 only)
__device__ __forceinline__ v16h load_A_text(const float* __restrict__ text, int row, int kh) {
  const f32x4* r4 = (const f32x4*)(text + (size_t)row * 16);
  v8h lo = cvt8(r4[kh * 2], r4[kh * 2 + 1]);
  v8h hi = {};
  hi[0] = kh ? (_Float16)0.0f : (_Float16)1.0f;   // k==16 bias column
  return cat16(lo, hi);
}

// image chunk0: k = 0..31, all real
__device__ __forceinline__ v16h load_A_img0(const float* __restrict__ image, int row, int kh) {
  const f32x4* r4 = (const f32x4*)(image + (size_t)row * 48);
  return cat16(cvt8(r4[kh * 2], r4[kh * 2 + 1]),
               cvt8(r4[4 + kh * 2], r4[5 + kh * 2]));
}

// image chunk1: k = 32..47 real, k==48 carries the bias-1
__device__ __forceinline__ v16h load_A_img1(const float* __restrict__ image, int row, int kh) {
  const f32x4* r4 = (const f32x4*)(image + (size_t)row * 48);
  v8h lo = cvt8(r4[8 + kh * 2], r4[9 + kh * 2]);
  v8h hi = {};
  hi[0] = kh ? (_Float16)0.0f : (_Float16)1.0f;   // k==48 bias column
  return cat16(lo, hi);
}

// A fragment from f16 activations in LDS, row length ld halves (32 or 64)
__device__ __forceinline__ v16h load_A_lds(const _Float16* h, int row0, int ld,
                                           int kchunk, int lane) {
  int m = lane & 15, kh = lane >> 4;
  const v8h* row = (const v8h*)(h + (row0 + m) * ld);
  return cat16(row[kchunk * 4 + kh], row[kchunk * 4 + 2 + kh]);
}

// ---------------- B fragment from transposed padded LDS weights Wt[n][k] ----------------
// lane&15 = N col (+16*ntile), lane>>4 = lh selects K range; k = 8lh..8lh+7 and 16+8lh..16+8lh+7 -> 2x b128
__device__ __forceinline__ v16h load_B_ldsT(const _Float16* Wt, int ldk,
                                            int ntile, int kchunk, int lane) {
  int n  = ntile * 16 + (lane & 15);
  int lh = lane >> 4;
  const v8h* p = (const v8h*)(Wt + n * ldk + kchunk * 32 + lh * 8);
  return cat16(p[0], p[2]);
}

// ================= quantum circuit pieces (per-thread, register state) =================
// amp index = q0*8 + q1*4 + q2*2 + q3 ; wire w <-> stride 8>>w

template <int STRIDE>
__device__ __forceinline__ void apply_ry(float c, float s, float* sr, float* si) {
#pragma unroll
  for (int p = 0; p < 8; ++p) {
    int lo = p & (STRIDE - 1);
    int i0 = ((p ^ lo) << 1) | lo;
    int i1 = i0 | STRIDE;
    float a0r = sr[i0], a0i = si[i0], a1r = sr[i1], a1i = si[i1];
    sr[i0] = c * a0r - s * a1r;  si[i0] = c * a0i - s * a1i;
    sr[i1] = s * a0r + c * a1r;  si[i1] = s * a0i + c * a1i;
  }
}

template <int STRIDE>
__device__ __forceinline__ void apply_rot(const float* g, float* sr, float* si) {
  float u00r = g[0], u00i = g[1], u01r = g[2], u01i = g[3];
  float u10r = g[4], u10i = g[5], u11r = g[6], u11i = g[7];
#pragma unroll
  for (int p = 0; p < 8; ++p) {
    int lo = p & (STRIDE - 1);
    int i0 = ((p ^ lo) << 1) | lo;
    int i1 = i0 | STRIDE;
    float a0r = sr[i0], a0i = si[i0], a1r = sr[i1], a1i = si[i1];
    sr[i0] = u00r * a0r - u00i * a0i + u01r * a1r - u01i * a1i;
    si[i0] = u00r * a0i + u00i * a0r + u01r * a1i + u01i * a1r;
    sr[i1] = u10r * a0r - u10i * a0i + u11r * a1r - u11i * a1i;
    si[i1] = u10r * a0i + u10i * a0r + u11r * a1i + u11i * a1r;
  }
}

// =======================================================================================

__global__ __launch_bounds__(QNN_BLOCK)
void QuantumNeuralNetwork_kernel(const float* __restrict__ text,  // [B,16]
                                 const float* __restrict__ image, // [B,48]
                                 const float* __restrict__ tW1,   // [16,32]
                                 const float* __restrict__ tb1,   // [32]
                                 const float* __restrict__ tW2,   // [32,4]
                                 const float* __restrict__ tb2,   // [4]
                                 const float* __restrict__ iW1,   // [48,64]
                                 const float* __restrict__ ib1,   // [64]
                                 const float* __restrict__ iW2,   // [64,4]
                                 const float* __restrict__ ib2,   // [4]
                                 const float* __restrict__ qw,    // [2,4,3] flat
                                 const float* __restrict__ cW1,   // [1,16]
                                 const float* __restrict__ cb1,   // [16]
                                 const float* __restrict__ cW2,   // [16,2]
                                 const float* __restrict__ cb2,   // [2]
                                 float* __restrict__ out)         // [B,2]
{
  __shared__ _Float16 h1t[QNN_BLOCK * 32];   // text hidden f16 (16 KB)
  __shared__ _Float16 h1i[QNN_BLOCK * 64];   // image hidden f16 (32 KB)
  __shared__ float    feats[QNN_BLOCK * 4];  // fused angles (4 KB)
  __shared__ float    gates[8 * 8];          // 8 Rot gates, complex 2x2 each
  // transposed, zero-padded, f16 weights: Wt[n][k]; layer-1 pad column holds bias
  __shared__ _Float16 wtW1t[32 * 32];        // n<32, k<32 (k==16 -> tb1)      2 KB
  __shared__ _Float16 wtW2t[16 * 32];        // n<16 (n>=4 pad), k<32          1 KB
  __shared__ _Float16 wiW1t[64 * 64];        // n<64, k<64 (k==48 -> ib1)      8 KB
  __shared__ _Float16 wiW2t[16 * 64];        // n<16 (n>=4 pad), k<64          2 KB

  const int tid  = threadIdx.x;
  const int lane = tid & 31;
  const int wv   = tid >> 5;
  const int blockbase = blockIdx.x * QNN_BLOCK;

  // ---- one-pass staging of padded transposed weights (no zero/fill race) ----
  for (int idx = tid; idx < 32 * 32; idx += QNN_BLOCK) {   // wtW1t
    int n = idx >> 5, k = idx & 31;
    float v = (k < 16) ? tW1[k * 32 + n] : (k == 16 ? tb1[n] : 0.0f);
    wtW1t[n * 32 + k] = (_Float16)v;
  }
  for (int idx = tid; idx < 16 * 32; idx += QNN_BLOCK) {   // wtW2t
    int n = idx >> 5, k = idx & 31;
    float v = (n < 4) ? tW2[k * 4 + n] : 0.0f;
    wtW2t[n * 32 + k] = (_Float16)v;
  }
  for (int idx = tid; idx < 64 * 64; idx += QNN_BLOCK) {   // wiW1t
    int n = idx >> 6, k = idx & 63;
    float v = (k < 48) ? iW1[k * 64 + n] : (k == 48 ? ib1[n] : 0.0f);
    wiW1t[n * 64 + k] = (_Float16)v;
  }
  for (int idx = tid; idx < 16 * 64; idx += QNN_BLOCK) {   // wiW2t
    int n = idx >> 6, k = idx & 63;
    float v = (n < 4) ? iW2[k * 4 + n] : 0.0f;
    wiW2t[n * 64 + k] = (_Float16)v;
  }

  // ---- the 8 uniform Rot matrices, once per block ----
  if (tid < 8) {
    float phi = qw[tid * 3 + 0], th = qw[tid * 3 + 1], om = qw[tid * 3 + 2];
    float ct = __cosf(0.5f * th), st = __sinf(0.5f * th);
    float a = 0.5f * (phi + om), b = 0.5f * (phi - om);
    float ca = __cosf(a), sa = __sinf(a), cb = __cosf(b), sb = __sinf(b);
    float* g = &gates[tid * 8];
    g[0] =  ca * ct;  g[1] = -sa * ct;   // u00 = e^{-i(phi+om)/2} cos(th/2)
    g[2] = -cb * st;  g[3] = -sb * st;   // u01 = -e^{+i(phi-om)/2} sin(th/2)
    g[4] =  cb * st;  g[5] = -sb * st;   // u10 = e^{-i(phi-om)/2} sin(th/2)
    g[6] =  ca * ct;  g[7] =  sa * ct;   // u11 = e^{+i(phi+om)/2} cos(th/2)
  }

  __syncthreads();   // weights + gates visible; the only block barrier

  const int m  = lane & 15;
  const int kh = lane >> 4;
  const int n4 = lane & 15;
  const float b2 = (n4 < 4) ? (tb2[n4] + ib2[n4]) : 0.0f;   // layer-2 biases

  // ================= WMMA feature pipeline: two 16-row M tiles per wave =================
#pragma unroll
  for (int mt = 0; mt < 2; ++mt) {
    const int r0l = wv * 32 + mt * 16;     // block-local sample base
    const int r0g = blockbase + r0l;       // global sample base

    // ---- text layer 1 (bias folded via k==16 column) ----
    v16h At = load_A_text(text, r0g + m, kh);
#pragma unroll
    for (int nt = 0; nt < 2; ++nt) {
      v16h Bt = load_B_ldsT(wtW1t, 32, nt, 0, lane);
      v8f acc = {};
      acc = wmma_f16(At, Bt, acc);
      int n = nt * 16 + (lane & 15);
#pragma unroll
      for (int r = 0; r < 8; ++r) {
        int smp = r0l + kh * 8 + r;                 // C layout: M = r + 8*(lane>=16)
        float hv = acc[r];
        h1t[smp * 32 + n] = (_Float16)(hv > 0.0f ? hv : 0.0f);
      }
    }

    // ---- image layer 1, 4 N-tiles x 2 K-chunks (bias via k==48 column) ----
    v16h Ai0 = load_A_img0(image, r0g + m, kh);
    v16h Ai1 = load_A_img1(image, r0g + m, kh);
#pragma unroll
    for (int nt = 0; nt < 4; ++nt) {
      v16h B0 = load_B_ldsT(wiW1t, 64, nt, 0, lane);
      v16h B1 = load_B_ldsT(wiW1t, 64, nt, 1, lane);
      v8f acc = {};
      acc = wmma_f16(Ai0, B0, acc);
      acc = wmma_f16(Ai1, B1, acc);
      int n = nt * 16 + (lane & 15);
#pragma unroll
      for (int r = 0; r < 8; ++r) {
        int smp = r0l + kh * 8 + r;
        float hv = acc[r];
        h1i[smp * 64 + n] = (_Float16)(hv > 0.0f ? hv : 0.0f);
      }
    }

    // intra-wave LDS transpose handoff: stores must land before A-fragment reads
    asm volatile("s_wait_dscnt 0" ::: "memory");

    // ---- text layer 2: [16x32] x [32x4 pad 16] ----
    v16h Ah  = load_A_lds(h1t, r0l, 32, 0, lane);
    v16h Bt2 = load_B_ldsT(wtW2t, 32, 0, 0, lane);
    v8f accT = {};
    accT = wmma_f16(Ah, Bt2, accT);

    // ---- image layer 2: [16x64] x [64x4 pad 16], 2 K-chunks ----
    v16h Ah0 = load_A_lds(h1i, r0l, 64, 0, lane);
    v16h Ah1 = load_A_lds(h1i, r0l, 64, 1, lane);
    v8f accI = {};
    accI = wmma_f16(Ah0, load_B_ldsT(wiW2t, 64, 0, 0, lane), accI);
    accI = wmma_f16(Ah1, load_B_ldsT(wiW2t, 64, 0, 1, lane), accI);

    // feats = 0.5 * (tf + imf + biases); only N columns 0..3 are real
    if (n4 < 4) {
#pragma unroll
      for (int r = 0; r < 8; ++r) {
        int smp = r0l + kh * 8 + r;
        feats[smp * 4 + n4] = 0.5f * (accT[r] + accI[r] + b2);
      }
    }
  }

  // feats written by this wave for its own 32 samples; wait, then simulate
  asm volatile("s_wait_dscnt 0" ::: "memory");

  // ================= per-thread 4-qubit statevector simulation =================
  f32x4 fv = *(const f32x4*)(feats + tid * 4);

  float sr[16], si[16];
#pragma unroll
  for (int i = 0; i < 16; ++i) { sr[i] = 0.0f; si[i] = 0.0f; }
  sr[0] = 1.0f;

  apply_ry<8>(__cosf(0.5f * fv.x), __sinf(0.5f * fv.x), sr, si);
  apply_ry<4>(__cosf(0.5f * fv.y), __sinf(0.5f * fv.y), sr, si);
  apply_ry<2>(__cosf(0.5f * fv.z), __sinf(0.5f * fv.z), sr, si);
  apply_ry<1>(__cosf(0.5f * fv.w), __sinf(0.5f * fv.w), sr, si);

#pragma unroll
  for (int l = 0; l < 2; ++l) {
    apply_rot<8>(&gates[(l * 4 + 0) * 8], sr, si);
    apply_rot<4>(&gates[(l * 4 + 1) * 8], sr, si);
    apply_rot<2>(&gates[(l * 4 + 2) * 8], sr, si);
    apply_rot<1>(&gates[(l * 4 + 3) * 8], sr, si);
#pragma unroll
    for (int idx = 0; idx < 16; ++idx) {
      int b0 = (idx >> 3) & 1, b1 = (idx >> 2) & 1, b2b = (idx >> 1) & 1, b3 = idx & 1;
      int par = (b0 & b1) ^ (b1 & b2b) ^ (b2b & b3);   // CZ chain combined sign
      if (par) { sr[idx] = -sr[idx]; si[idx] = -si[idx]; }
    }
  }

  float q = 0.0f;
#pragma unroll
  for (int idx = 0; idx < 16; ++idx) {
    float mag = sr[idx] * sr[idx] + si[idx] * si[idx];
    q += (idx < 8) ? mag : -mag;                       // <Z_0>
  }

  // ================= rank-1 classifier head =================
  float o0 = cb2[0], o1 = cb2[1];
#pragma unroll
  for (int j = 0; j < 16; ++j) {
    float h = q * cW1[j] + cb1[j];
    h = h > 0.0f ? h : 0.0f;
    o0 += h * cW2[j * 2 + 0];
    o1 += h * cW2[j * 2 + 1];
  }

  size_t s = (size_t)(blockbase + tid);
  float2 res; res.x = o0; res.y = o1;
  *(float2*)(out + s * 2) = res;
}

extern "C" void kernel_launch(void* const* d_in, const int* in_sizes, int n_in,
                              void* d_out, int out_size, void* d_ws, size_t ws_size,
                              hipStream_t stream) {
  const float* text  = (const float*)d_in[0];
  const float* image = (const float*)d_in[1];
  const float* tW1   = (const float*)d_in[2];
  const float* tb1   = (const float*)d_in[3];
  const float* tW2   = (const float*)d_in[4];
  const float* tb2   = (const float*)d_in[5];
  const float* iW1   = (const float*)d_in[6];
  const float* ib1   = (const float*)d_in[7];
  const float* iW2   = (const float*)d_in[8];
  const float* ib2   = (const float*)d_in[9];
  const float* qw    = (const float*)d_in[10];
  const float* cW1   = (const float*)d_in[11];
  const float* cb1   = (const float*)d_in[12];
  const float* cW2   = (const float*)d_in[13];
  const float* cb2   = (const float*)d_in[14];
  float* out = (float*)d_out;

  const int batch  = in_sizes[0] / 16;          // 524288
  const int blocks = batch / QNN_BLOCK;         // 2048 (exact)

  QuantumNeuralNetwork_kernel<<<blocks, QNN_BLOCK, 0, stream>>>(
      text, image, tW1, tb1, tW2, tb2, iW1, ib1, iW2, ib2,
      qw, cW1, cb1, cW2, cb2, out);
}
